// ReformerEncoder_51479478010644
// MI455X (gfx1250) — matrix operand order repaired and checked by reading, not compiled, not run on hardware
//
#include <hip/hip_runtime.h>
#include <hip/hip_bf16.h>
#include <math.h>

// ---------------- model constants ----------------
constexpr int B  = 4;
constexpr int S  = 4096;
constexpr int D  = 512;
constexpr int H  = 8;
constexpr int L  = 6;
constexpr int M  = 2048;
constexpr int C  = 64;         // chunk length
constexpr int NC = S / C;      // 64 chunks
constexpr int DH = D / H;      // 64
constexpr int NBH = 32;        // NB/2 rotation projections
constexpr int BS = B * S;      // 16384 rows

typedef __bf16 bf16_t;
typedef bf16_t v16bf __attribute__((ext_vector_type(16)));
typedef bf16_t v8bf  __attribute__((ext_vector_type(8)));
typedef float  v8f   __attribute__((ext_vector_type(8)));
typedef float  v4f   __attribute__((ext_vector_type(4)));
typedef unsigned u32x4 __attribute__((ext_vector_type(4)));
typedef int    i32x8 __attribute__((ext_vector_type(8)));
typedef int    i32x4 __attribute__((ext_vector_type(4)));

#if __has_builtin(__builtin_amdgcn_tensor_load_to_lds) && \
    __has_builtin(__builtin_amdgcn_s_wait_tensorcnt)
#define USE_TDM 1
#else
#define USE_TDM 0
#endif

__device__ inline bf16_t f2bf(float f) { return (bf16_t)f; }

__device__ inline v16bf cat8(v8bf lo, v8bf hi) {
  return __builtin_shufflevector(lo, hi, 0, 1, 2, 3, 4, 5, 6, 7, 8, 9, 10, 11,
                                 12, 13, 14, 15);
}

__device__ inline v8f wmma_bf16(v16bf a, v16bf b, v8f c) {
  // D = A(16x32 bf16) * B(32x16 bf16) + C(f32)
  return __builtin_amdgcn_wmma_f32_16x16x32_bf16(false, a, false, b, (short)0, c,
                                                 false, false);
}

__device__ inline float gelu_tanh(float x) {
  float x3 = x * x * x;
  return 0.5f * x * (1.f + tanhf(0.7978845608028654f * (x + 0.044715f * x3)));
}

#if USE_TDM
// Issue a 2D TDM load: tile (32 x tileRows) of bf16 from a [rows x K] tensor
// into LDS at byte offset ldsOff. pad_enable inserts 4 DWORDs (8 bf16) after
// every 16 DWORDs (32 bf16 = one tile row), producing LDS row stride 40 bf16.
__device__ inline void tdm_issue_2d(unsigned ldsOff, const bf16_t* g, int K_,
                                    int rows, int tileRows) {
  unsigned long long ga = (unsigned long long)(uintptr_t)g;
  u32x4 g0;
  g0.x = 1u;                                  // count=1 (valid descriptor)
  g0.y = ldsOff;                              // lds_addr
  g0.z = (unsigned)ga;                        // global_addr[31:0]
  g0.w = (unsigned)((ga >> 32) & 0x01ffffffull) | (2u << 30);  // addr hi, type=2
  i32x8 g1;
  g1[0] = (int)((1u << 16) |                  // data_size = 2 bytes
                (1u << 20) |                  // pad_enable
                (3u << 22) |                  // pad_interval: 16 DWORDs
                (3u << 25));                  // pad_amount: 4 DWORDs
  g1[1] = (int)(((unsigned)K_ & 0xffffu) << 16);           // tensor_dim0 lo
  g1[2] = (int)(((unsigned)K_ >> 16) |
                (((unsigned)rows & 0xffffu) << 16));       // dim0 hi, dim1 lo
  g1[3] = (int)((((unsigned)rows >> 16) & 0xffffu) |
                (32u << 16));                              // dim1 hi, tile_dim0=32
  g1[4] = tileRows;                                        // tile_dim1
  g1[5] = K_;                                              // tensor_dim0_stride
  g1[6] = 0;
  g1[7] = 0;
  i32x4 z4 = (i32x4)0;
#if __clang_major__ >= 23
  __builtin_amdgcn_tensor_load_to_lds(g0, g1, z4, z4, (i32x8)0, 0);
#else
  __builtin_amdgcn_tensor_load_to_lds(g0, g1, z4, z4, 0);
#endif
}
#endif

// ---------------- embedding + sinusoid positional encoding ----------------
__global__ __launch_bounds__(256) void embed_kernel(
    const int* __restrict__ tokens, const float* __restrict__ emb,
    float* __restrict__ x) {
  long g = (long)blockIdx.x * 256 + threadIdx.x;
  if (g >= (long)BS * D) return;
  int d = (int)(g % D);
  long bs = g / D;
  int s = (int)(bs % S);
  int t = tokens[bs];
  // freq(i) = exp(-i * ln(10000)/255)
  const float NEG = -9.210340371976184f / 255.f;
  float pe;
  if (d < 256) pe = __sinf((float)s * __expf((float)d * NEG));
  else         pe = __cosf((float)s * __expf((float)(d - 256) * NEG));
  x[g] = emb[(size_t)t * D + d] + pe;
}

// ---------------- layernorm (row = one block of 256 threads, D=512) -------
__global__ __launch_bounds__(256) void ln_kernel(
    const float* __restrict__ x, const float* __restrict__ sc,
    const float* __restrict__ bi, float* __restrict__ outf,
    bf16_t* __restrict__ outb) {
  __shared__ float red[256];
  int row = blockIdx.x, tid = threadIdx.x;
  const float* xr = x + (size_t)row * D;
  float a0 = xr[tid], a1 = xr[tid + 256];
  red[tid] = a0 + a1;
  __syncthreads();
  for (int s = 128; s > 0; s >>= 1) {
    if (tid < s) red[tid] += red[tid + s];
    __syncthreads();
  }
  float mean = red[0] * (1.f / D);
  __syncthreads();
  float d0 = a0 - mean, d1 = a1 - mean;
  red[tid] = d0 * d0 + d1 * d1;
  __syncthreads();
  for (int s = 128; s > 0; s >>= 1) {
    if (tid < s) red[tid] += red[tid + s];
    __syncthreads();
  }
  float rstd = rsqrtf(red[0] * (1.f / D) + 1e-6f);
  float o0 = d0 * rstd * sc[tid] + bi[tid];
  float o1 = d1 * rstd * sc[tid + 256] + bi[tid + 256];
  size_t base = (size_t)row * D;
  if (outf) { outf[base + tid] = o0; outf[base + tid + 256] = o1; }
  if (outb) { outb[base + tid] = f2bf(o0); outb[base + tid + 256] = f2bf(o1); }
}

// ---- weight convert + transpose: src[K][N] f32 -> dst[N][K] bf16 ----------
__global__ __launch_bounds__(256) void convT_kernel(
    const float* __restrict__ src, bf16_t* __restrict__ dst, int K, int N) {
  __shared__ float tile[32][33];
  int k0 = blockIdx.x * 32;
  int n0 = blockIdx.y * 32;
  int tr = threadIdx.x >> 5;   // 0..7
  int tc = threadIdx.x & 31;
  for (int rr = tr; rr < 32; rr += 8)
    tile[rr][tc] = src[(size_t)(k0 + rr) * N + n0 + tc];
  __syncthreads();
  for (int rr = tr; rr < 32; rr += 8)
    dst[(size_t)(n0 + rr) * K + k0 + tc] = f2bf(tile[tc][rr]);
}

// ---------------- bf16 WMMA GEMM: C = A[Mtot,K] * Bt[N,K]^T ---------------
// Block: 256 threads (8 waves), 128x64 output tile, K-step 32, double-
// buffered LDS. Tile staging goes through the Tensor Data Mover when
// available (one wave issues two descriptors per step; prefetch depth 1;
// s_wait_tensorcnt gates the consumer), else a register-prefetch pipeline.
// Every WMMA fragment is two K-contiguous 16B ds_load_b128 per lane.
constexpr int LSTR = 40;  // padded LDS stride in bf16 (80B, 16B-aligned)
__global__ __launch_bounds__(256) void gemm_bf16_kernel(
    const bf16_t* __restrict__ A, const bf16_t* __restrict__ Bt,
    float* __restrict__ Cf, bf16_t* __restrict__ Cbf,
    const float* __restrict__ bias, const float* __restrict__ res,
    int N, int K, int gelu) {
  __shared__ __align__(16) bf16_t As[2 * 128 * LSTR];
  __shared__ __align__(16) bf16_t Bs[2 * 64 * LSTR];
  int tid = threadIdx.x;
  int lane = tid & 31;
  int wave = tid >> 5;
  int bm = blockIdx.x * 128;
  int bn = blockIdx.y * 64;
  int mtB = (wave & 3) * 2;   // tile-row base: 0,2,4,6
  int ntB = (wave >> 2) * 2;  // tile-col base: 0,2
  int kbA = (lane & 16) ? 8 : 0;   // A frag: runs at kbA and 16+kbA
  int kbB = (lane & 16) ? 8 : 0;   // B frag: runs at kbB and 16+kbB
  int l15 = lane & 15;
  int ar0 = (mtB + 0) * 16 + l15;
  int ar1 = (mtB + 1) * 16 + l15;
  int bc0 = (ntB + 0) * 16 + l15;
  int bc1 = (ntB + 1) * 16 + l15;
  v8f acc00 = {}, acc01 = {}, acc10 = {}, acc11 = {};
  int nsteps = K / 32;

#if USE_TDM
  unsigned asOff = (unsigned)(uintptr_t)(void*)&As[0];
  unsigned bsOff = (unsigned)(uintptr_t)(void*)&Bs[0];
  const bf16_t* aTile = A + (size_t)bm * K;
  const bf16_t* bTile = Bt + (size_t)bn * K;
  if (wave == 0) {
    tdm_issue_2d(asOff, aTile, K, 1 << 20, 128);
    tdm_issue_2d(bsOff, bTile, K, N, 64);
  }
#else
  // register-prefetch pipeline coordinates: r = tid>>2, c8 = (tid&3)*8
  int cr = tid >> 2, cc = (tid & 3) * 8;
  const bf16_t* aP0 = A + (size_t)(bm + cr) * K + cc;
  const bf16_t* aP1 = A + (size_t)(bm + cr + 64) * K + cc;
  const bf16_t* bP = Bt + (size_t)(bn + cr) * K + cc;
  v8bf ra0 = *(const v8bf*)aP0;
  v8bf ra1 = *(const v8bf*)aP1;
  v8bf rb = *(const v8bf*)bP;
#endif

  for (int ks = 0; ks < nsteps; ++ks) {
    int cur = ks & 1;
    const bf16_t* Ac = As + cur * 128 * LSTR;
    const bf16_t* Bc = Bs + cur * 64 * LSTR;
#if USE_TDM
    if (wave == 0) {
      if (ks + 1 < nsteps) {
        int nxt = (ks + 1) & 1;
        tdm_issue_2d(asOff + (unsigned)(nxt * 128 * LSTR * 2),
                     aTile + (ks + 1) * 32, K, 1 << 20, 128);
        tdm_issue_2d(bsOff + (unsigned)(nxt * 64 * LSTR * 2),
                     bTile + (ks + 1) * 32, K, N, 64);
        __builtin_amdgcn_s_wait_tensorcnt(2);  // current pair complete
      } else {
        __builtin_amdgcn_s_wait_tensorcnt(0);
      }
    }
    __syncthreads();
#else
    bf16_t* Aw = (bf16_t*)Ac;
    bf16_t* Bw = (bf16_t*)Bc;
    *(v8bf*)&Aw[cr * LSTR + cc] = ra0;
    *(v8bf*)&Aw[(cr + 64) * LSTR + cc] = ra1;
    *(v8bf*)&Bw[cr * LSTR + cc] = rb;
    __syncthreads();
    if (ks + 1 < nsteps) {
      aP0 += 32; aP1 += 32; bP += 32;
      ra0 = *(const v8bf*)aP0;
      ra1 = *(const v8bf*)aP1;
      rb = *(const v8bf*)bP;
    }
#endif
    v16bf a0 = cat8(*(const v8bf*)&Ac[ar0 * LSTR + kbA],
                    *(const v8bf*)&Ac[ar0 * LSTR + 16 + kbA]);
    v16bf a1 = cat8(*(const v8bf*)&Ac[ar1 * LSTR + kbA],
                    *(const v8bf*)&Ac[ar1 * LSTR + 16 + kbA]);
    v16bf b0 = cat8(*(const v8bf*)&Bc[bc0 * LSTR + kbB],
                    *(const v8bf*)&Bc[bc0 * LSTR + kbB + 16]);
    v16bf b1 = cat8(*(const v8bf*)&Bc[bc1 * LSTR + kbB],
                    *(const v8bf*)&Bc[bc1 * LSTR + kbB + 16]);
    acc00 = wmma_bf16(a0, b0, acc00);
    acc01 = wmma_bf16(a0, b1, acc01);
    acc10 = wmma_bf16(a1, b0, acc10);
    acc11 = wmma_bf16(a1, b1, acc11);
    __syncthreads();
  }
  // epilogue: 4 tiles
#pragma unroll
  for (int q = 0; q < 4; ++q) {
    v8f acc = (q == 0) ? acc00 : (q == 1) ? acc01 : (q == 2) ? acc10 : acc11;
    int mt = mtB + (q >> 1);
    int gcol = bn + ((q & 1) ? bc1 : bc0);
    int rbase = bm + mt * 16 + ((lane & 16) ? 8 : 0);
    float bval = bias ? bias[gcol] : 0.f;
#pragma unroll
    for (int r = 0; r < 8; ++r) {
      size_t idx = (size_t)(rbase + r) * N + gcol;
      float v = acc[r] + bval;
      if (gelu) v = gelu_tanh(v);
      if (res) v += res[idx];
      if (Cf) Cf[idx] = v;
      if (Cbf) Cbf[idx] = f2bf(v);
    }
  }
}

// ---------------- LSH hashing: bucket = argmax([r,-r]), key = bucket*S + s --
__global__ __launch_bounds__(256) void hash_kernel(
    const float* __restrict__ qbuf, const float* __restrict__ rot_l,
    unsigned* __restrict__ keys) {
  int g = blockIdx.x * 256 + threadIdx.x;   // over B*H*S
  if (g >= B * H * S) return;
  int s = g % S;
  int h = (g / S) % H;
  int b = g / (S * H);
  const float* qrow = qbuf + ((size_t)b * S + s) * D + h * DH;
  const float* rr = rot_l + (size_t)h * DH * NBH;
  float qv[DH];
#pragma unroll
  for (int f = 0; f < DH; ++f) qv[f] = qrow[f];
  float best = -3.4e38f;
  int bi = 0;
  for (int j = 0; j < NBH; ++j) {
    float acc = 0.f;
#pragma unroll
    for (int f = 0; f < DH; ++f) acc += qv[f] * rr[f * NBH + j];
    if (acc > best) { best = acc; bi = j; }
    if (-acc > best) { best = -acc; bi = NBH + j; }
  }
  keys[g] = (unsigned)bi * (unsigned)S + (unsigned)s;
}

// ---------------- per-(b,h) bitonic argsort of 4096 keys in LDS ----------
__global__ __launch_bounds__(1024) void sort_kernel(
    const unsigned* __restrict__ keys, unsigned* __restrict__ st) {
  __shared__ unsigned skey[S];
  __shared__ unsigned sval[S];
  int bh = blockIdx.x;
  int tid = threadIdx.x;
  for (int i = tid; i < S; i += 1024) {
    skey[i] = keys[(size_t)bh * S + i];
    sval[i] = (unsigned)i;
  }
  for (unsigned k = 2; k <= (unsigned)S; k <<= 1) {
    for (unsigned j = k >> 1; j > 0; j >>= 1) {
      __syncthreads();
      for (unsigned i = tid; i < (unsigned)S; i += 1024) {
        unsigned ixj = i ^ j;
        if (ixj > i) {
          bool up = ((i & k) == 0);
          unsigned ki = skey[i], kj = skey[ixj];
          if ((ki > kj) == up) {
            skey[i] = kj; skey[ixj] = ki;
            unsigned vi = sval[i]; sval[i] = sval[ixj]; sval[ixj] = vi;
          }
        }
      }
    }
  }
  __syncthreads();
  for (int i = tid; i < S; i += 1024) st[(size_t)bh * S + i] = sval[i];
}

// ---------------- gather sorted q/v (bf16) + pad mask (8 elems/thread) ----
__global__ __launch_bounds__(256) void gather_kernel(
    const float* __restrict__ qbuf, const bf16_t* __restrict__ vbuf,
    const int* __restrict__ tokens, const unsigned* __restrict__ st,
    bf16_t* __restrict__ sq, bf16_t* __restrict__ sv,
    float* __restrict__ spad) {
  long g = (long)blockIdx.x * 256 + threadIdx.x;  // over B*H*S*(DH/8)
  if (g >= (long)B * H * S * (DH / 8)) return;
  int f8 = (int)(g & 7) * 8;
  long bhj = g >> 3;
  int j = (int)(bhj % S);
  int bh = (int)(bhj / S);
  int h = bh % H, b = bh / H;
  unsigned ssrc = st[(size_t)bh * S + j];
  size_t src = ((size_t)b * S + ssrc) * D + h * DH + f8;
  v4f q0 = *(const v4f*)&qbuf[src];
  v4f q1 = *(const v4f*)&qbuf[src + 4];
  v8bf qb;
#pragma unroll
  for (int e = 0; e < 4; ++e) { qb[e] = f2bf(q0[e]); qb[4 + e] = f2bf(q1[e]); }
  size_t dst = (size_t)bhj * DH + f8;
  *(v8bf*)&sq[dst] = qb;
  *(v8bf*)&sv[dst] = *(const v8bf*)&vbuf[src];
  if (f8 == 0)
    spad[(size_t)bh * S + j] = (tokens[(size_t)b * S + ssrc] > 0) ? 1.f : 0.f;
}

// ---------------- chunked LSH attention (one chunk per block) ------------
constexpr int VSTR = 136;  // padded transposed-V stride (272B, 16B-aligned)
__global__ __launch_bounds__(256) void attn_kernel(
    const bf16_t* __restrict__ sq, const bf16_t* __restrict__ sv,
    const float* __restrict__ spad, const unsigned* __restrict__ st,
    bf16_t* __restrict__ obuf) {
  // LDS layout (bytes):
  //   [0,17408)      Vt  transposed V, 64 x VSTR bf16
  //   [17408,25600)  Q    64x64 bf16     \ overlaid later by logits
  //   [25600,41984)  K   128x64 bf16     / 64x128 f32 (17408..50176)
  //   logits rows compacted in-place into bf16 P rows (stride 256 bf16)
  //   [50176,50688)  key pad mask, 128 f32
  __shared__ __align__(16) unsigned char smem[50688];
  bf16_t* Vtt = (bf16_t*)smem;
  bf16_t* Qt = (bf16_t*)(smem + 17408);
  bf16_t* Kt = (bf16_t*)(smem + 25600);
  float* Lg = (float*)(smem + 17408);   // [64][128]
  bf16_t* P = (bf16_t*)(smem + 17408);  // row stride 256 (overlay of Lg)
  float* Msk = (float*)(smem + 50176);  // [128]

  int tid = threadIdx.x;
  int lane = tid & 31;
  int w = tid >> 5;
  int n = blockIdx.x % NC;
  int h = (blockIdx.x / NC) % H;
  int b = blockIdx.x / (NC * H);
  int nPrev = (n + NC - 1) % NC;
  size_t base = (size_t)(b * H + h) * S;
  const bf16_t* sqb = sq + base * DH;
  const bf16_t* svb = sv + base * DH;

  for (int i = tid; i < 64 * 8; i += 256) {  // 64 rows x 8 groups of 8
    int r = i >> 3, c8 = (i & 7) * 8;
    v8bf qv = *(const v8bf*)&sqb[(size_t)(n * C + r) * DH + c8];
    v8bf qp = *(const v8bf*)&sqb[(size_t)(nPrev * C + r) * DH + c8];
    *(v8bf*)&Qt[r * 64 + c8] = qv;          // shared QK
    *(v8bf*)&Kt[r * 64 + c8] = qv;
    *(v8bf*)&Kt[(64 + r) * 64 + c8] = qp;
    v8bf vv = *(const v8bf*)&svb[(size_t)(n * C + r) * DH + c8];
    v8bf vp = *(const v8bf*)&svb[(size_t)(nPrev * C + r) * DH + c8];
#pragma unroll
    for (int e = 0; e < 8; ++e) {
      Vtt[(c8 + e) * VSTR + r] = vv[e];
      Vtt[(c8 + e) * VSTR + 64 + r] = vp[e];
    }
  }
  if (tid < 128) {
    int cc = (tid < 64) ? (n * C + tid) : (nPrev * C + (tid - 64));
    Msk[tid] = spad[base + cc];
  }
  __syncthreads();

  // ---- phase 1: logits = Q * K^T  (M=64, N=128, K=64); wave w owns n-tile w
  int kbA = (lane & 16) ? 8 : 0;
  int kbB = (lane & 16) ? 8 : 0;
  v8f accL[4] = {v8f{}, v8f{}, v8f{}, v8f{}};
  {
    int col = w * 16 + (lane & 15);   // key index (N dim)
    v16bf b0 = cat8(*(const v8bf*)&Kt[col * 64 + kbB],
                    *(const v8bf*)&Kt[col * 64 + kbB + 16]);
    v16bf b1 = cat8(*(const v8bf*)&Kt[col * 64 + 32 + kbB],
                    *(const v8bf*)&Kt[col * 64 + 32 + kbB + 16]);
#pragma unroll
    for (int mt = 0; mt < 4; ++mt) {
      int row = mt * 16 + (lane & 15);
      v16bf a0 = cat8(*(const v8bf*)&Qt[row * 64 + kbA],
                      *(const v8bf*)&Qt[row * 64 + 16 + kbA]);
      v16bf a1 = cat8(*(const v8bf*)&Qt[row * 64 + 32 + kbA],
                      *(const v8bf*)&Qt[row * 64 + 48 + kbA]);
      accL[mt] = wmma_bf16(a0, b0, accL[mt]);
      accL[mt] = wmma_bf16(a1, b1, accL[mt]);
    }
  }
  __syncthreads();  // everyone done reading Q/K before logits overlay
#pragma unroll
  for (int mt = 0; mt < 4; ++mt) {
    int rb = mt * 16 + ((lane & 16) ? 8 : 0);
    int col = w * 16 + (lane & 15);
#pragma unroll
    for (int r = 0; r < 8; ++r) Lg[(rb + r) * 128 + col] = accL[mt][r];
  }
  __syncthreads();

  // ---- phase 2: masked softmax (thread tid<64 owns row tid), write P (bf16)
  if (tid < 64) {
    float mx = -1e30f;
    for (int c = 0; c < 128; ++c) {
      float l = Lg[tid * 128 + c] * 0.125f;       // 1/sqrt(DH)
      l = (Msk[c] != 0.f) ? l : -1e9f;
      Lg[tid * 128 + c] = l;
      mx = fmaxf(mx, l);
    }
    float sum = 0.f;
    for (int c = 0; c < 128; ++c) {
      float e = __expf(Lg[tid * 128 + c] - mx);
      Lg[tid * 128 + c] = e;
      sum += e;
    }
    float inv = 1.f / sum;
    bf16_t* prow = P + tid * 256;
    for (int c = 0; c < 128; ++c) prow[c] = f2bf(Lg[tid * 128 + c] * inv);
  }
  __syncthreads();

  // ---- phase 3: O = P(64x128) * V(128x64); wave w owns tiles 2w, 2w+1
  int t0 = 2 * w;
  int mt = t0 >> 2;
  int nt0 = t0 & 3, nt1 = (t0 + 1) & 3;
  int arow = mt * 16 + (lane & 15);
  int bc0 = nt0 * 16 + (lane & 15);
  int bc1 = nt1 * 16 + (lane & 15);
  v8f accO0 = {}, accO1 = {};
#pragma unroll
  for (int ks = 0; ks < 4; ++ks) {
    int k0 = ks * 32;
    v16bf a = cat8(*(const v8bf*)&P[arow * 256 + k0 + kbA],
                   *(const v8bf*)&P[arow * 256 + k0 + 16 + kbA]);
    v16bf b0 = cat8(*(const v8bf*)&Vtt[bc0 * VSTR + k0 + kbB],
                    *(const v8bf*)&Vtt[bc0 * VSTR + k0 + kbB + 16]);
    v16bf b1 = cat8(*(const v8bf*)&Vtt[bc1 * VSTR + k0 + kbB],
                    *(const v8bf*)&Vtt[bc1 * VSTR + k0 + kbB + 16]);
    accO0 = wmma_bf16(a, b0, accO0);
    accO1 = wmma_bf16(a, b1, accO1);
  }
  // scatter back to original token order via st
  const unsigned* stb = st + base;
  int rb = mt * 16 + ((lane & 16) ? 8 : 0);
  for (int half = 0; half < 2; ++half) {
    v8f acc = half ? accO1 : accO0;
    int col = half ? bc1 : bc0;
#pragma unroll
    for (int r = 0; r < 8; ++r) {
      int j = n * C + rb + r;
      unsigned sorig = stb[j];
      obuf[((size_t)b * S + sorig) * D + h * DH + col] = f2bf(acc[r]);
    }
  }
}

// ---------------- host driver ----------------
static inline size_t align256(size_t x) { return (x + 255) & ~(size_t)255; }

extern "C" void kernel_launch(void* const* d_in, const int* in_sizes, int n_in,
                              void* d_out, int out_size, void* d_ws,
                              size_t ws_size, hipStream_t stream) {
  (void)in_sizes; (void)n_in; (void)out_size; (void)ws_size;
  const int* tokens = (const int*)d_in[0];
  const float* emb = (const float*)d_in[1];
  const float* rot = (const float*)d_in[2];
  const float* ln1_s = (const float*)d_in[3];
  const float* ln1_b = (const float*)d_in[4];
  const float* Wq = (const float*)d_in[5];
  const float* Wv = (const float*)d_in[6];
  const float* Wo = (const float*)d_in[7];
  const float* ln2_s = (const float*)d_in[8];
  const float* ln2_b = (const float*)d_in[9];
  const float* W1 = (const float*)d_in[10];
  const float* b1 = (const float*)d_in[11];
  const float* W2 = (const float*)d_in[12];
  const float* b2 = (const float*)d_in[13];
  const float* lnf_s = (const float*)d_in[14];
  const float* lnf_b = (const float*)d_in[15];
  float* out = (float*)d_out;

  // workspace carve-out (~222 MB)
  char* ws = (char*)d_ws;
  size_t off = 0;
  auto carve = [&](size_t bytes) {
    void* p = ws + off;
    off = align256(off + bytes);
    return p;
  };
  float* x = (float*)carve((size_t)BS * D * 4);
  float* qbuf = (float*)carve((size_t)BS * D * 4);
  bf16_t* h_bf = (bf16_t*)carve((size_t)BS * D * 2);
  bf16_t* v_bf = (bf16_t*)carve((size_t)BS * D * 2);
  bf16_t* o_bf = (bf16_t*)carve((size_t)BS * D * 2);
  bf16_t* u_bf = (bf16_t*)carve((size_t)BS * M * 2);
  bf16_t* w_bf = (bf16_t*)carve((size_t)D * M * 2);  // transposed weights
  unsigned* keys = (unsigned*)carve((size_t)B * H * S * 4);
  unsigned* st = (unsigned*)carve((size_t)B * H * S * 4);
  bf16_t* sq_bf = (bf16_t*)carve((size_t)B * H * S * DH * 2);
  bf16_t* sv_bf = (bf16_t*)carve((size_t)B * H * S * DH * 2);
  float* spad = (float*)carve((size_t)B * H * S * 4);

  const long nXD = (long)BS * D;
  const int gXD = (int)((nXD + 255) / 256);

  embed_kernel<<<gXD, 256, 0, stream>>>(tokens, emb, x);

  for (int l = 0; l < L; ++l) {
    const float* Wq_l = Wq + (size_t)l * D * D;
    const float* Wv_l = Wv + (size_t)l * D * D;
    const float* Wo_l = Wo + (size_t)l * D * D;
    const float* W1_l = W1 + (size_t)l * D * M;
    const float* W2_l = W2 + (size_t)l * M * D;
    const float* rot_l = rot + (size_t)l * H * DH * NBH;

    // LN1 -> bf16 activations
    ln_kernel<<<BS, 256, 0, stream>>>(x, ln1_s + (size_t)l * D,
                                      ln1_b + (size_t)l * D, nullptr, h_bf);
    // Q = h @ Wq (f32 out, needed for hashing)
    convT_kernel<<<dim3(D / 32, D / 32), 256, 0, stream>>>(Wq_l, w_bf, D, D);
    gemm_bf16_kernel<<<dim3(BS / 128, D / 64), 256, 0, stream>>>(
        h_bf, w_bf, qbuf, nullptr, nullptr, nullptr, D, D, 0);
    // V = h @ Wv (bf16 out)
    convT_kernel<<<dim3(D / 32, D / 32), 256, 0, stream>>>(Wv_l, w_bf, D, D);
    gemm_bf16_kernel<<<dim3(BS / 128, D / 64), 256, 0, stream>>>(
        h_bf, w_bf, nullptr, v_bf, nullptr, nullptr, D, D, 0);
    // LSH bucket hash -> sort keys
    hash_kernel<<<(B * H * S) / 256, 256, 0, stream>>>(qbuf, rot_l, keys);
    // stable argsort per (b,h) via bitonic sort (keys are unique)
    sort_kernel<<<B * H, 1024, 0, stream>>>(keys, st);
    // gather sorted q/v + pad mask
    gather_kernel<<<(int)(((long)B * H * S * (DH / 8) + 255) / 256), 256, 0,
                    stream>>>(qbuf, v_bf, tokens, st, sq_bf, sv_bf, spad);
    // chunked attention (WMMA), scatter to original order as bf16
    attn_kernel<<<B * H * NC, 256, 0, stream>>>(sq_bf, sv_bf, spad, st, o_bf);
    // x += attn_out @ Wo
    convT_kernel<<<dim3(D / 32, D / 32), 256, 0, stream>>>(Wo_l, w_bf, D, D);
    gemm_bf16_kernel<<<dim3(BS / 128, D / 64), 256, 0, stream>>>(
        o_bf, w_bf, x, nullptr, nullptr, x, D, D, 0);
    // LN2 -> bf16
    ln_kernel<<<BS, 256, 0, stream>>>(x, ln2_s + (size_t)l * D,
                                      ln2_b + (size_t)l * D, nullptr, h_bf);
    // u = gelu(h @ W1 + b1) (bf16 out)
    convT_kernel<<<dim3(D / 32, M / 32), 256, 0, stream>>>(W1_l, w_bf, D, M);
    gemm_bf16_kernel<<<dim3(BS / 128, M / 64), 256, 0, stream>>>(
        h_bf, w_bf, nullptr, u_bf, b1 + (size_t)l * M, nullptr, M, D, 1);
    // x += u @ W2 + b2
    convT_kernel<<<dim3(M / 32, D / 32), 256, 0, stream>>>(W2_l, w_bf, M, D);
    gemm_bf16_kernel<<<dim3(BS / 128, D / 64), 256, 0, stream>>>(
        u_bf, w_bf, x, nullptr, b2 + (size_t)l * D, x, D, M, 0);
  }
  // final layernorm -> output (f32)
  ln_kernel<<<BS, 256, 0, stream>>>(x, lnf_s, lnf_b, out, nullptr);
}
